// SimpleGNNModel_64639257805082
// MI455X (gfx1250) — matrix operand order repaired and checked by reading, not compiled, not run on hardware
//
#include <hip/hip_runtime.h>

typedef __attribute__((ext_vector_type(16))) _Float16 v16h;
typedef __attribute__((ext_vector_type(8)))  _Float16 v8h;
typedef __attribute__((ext_vector_type(8)))  float    v8f;

#define F_DIM 128

// ---------------------------------------------------------------------------
// degree / normalization helpers
// ---------------------------------------------------------------------------
__global__ __launch_bounds__(256) void init_deg(float* __restrict__ deg, int n) {
    int i = blockIdx.x * blockDim.x + threadIdx.x;
    if (i < n) deg[i] = 1.0f;               // self-loop contributes 1
}

__global__ __launch_bounds__(256) void edge_deg(const int* __restrict__ dst,
                                                float* __restrict__ deg, int e) {
    int i = blockIdx.x * blockDim.x + threadIdx.x;
    if (i < e) unsafeAtomicAdd(&deg[dst[i]], 1.0f);
}

__global__ __launch_bounds__(256) void make_dis(float* __restrict__ deg_dis, int n) {
    int i = blockIdx.x * blockDim.x + threadIdx.x;
    if (i < n) {
        float d = deg_dis[i];
        deg_dis[i] = d > 0.0f ? rsqrtf(d) : 0.0f;   // in-place deg -> dis
    }
}

// per-edge norm, computed once and reused by all 3 layers
__global__ __launch_bounds__(256)
void make_norm(const int* __restrict__ src, const int* __restrict__ dst,
               const float* __restrict__ dis, float* __restrict__ normE, int e) {
    int i = blockIdx.x * blockDim.x + threadIdx.x;
    if (i < e) normE[i] = dis[src[i]] * dis[dst[i]];
}

// ---------------------------------------------------------------------------
// Pre-convert W (128x128, row-major [k][c]) into TRANSPOSED f16 hi/lo arrays
// Wt[c][k] so the WMMA B fragment is a contiguous 32-byte load per K-step.
// ---------------------------------------------------------------------------
__global__ __launch_bounds__(256)
void conv_w(const float* __restrict__ W, _Float16* __restrict__ Wh,
            _Float16* __restrict__ Wl) {
    int t = blockIdx.x * blockDim.x + threadIdx.x;
    if (t >= F_DIM * F_DIM) return;
    int k = t >> 7, c = t & (F_DIM - 1);
    float v = W[t];                       // W[k][c]
    _Float16 h = (_Float16)v;
    Wh[(size_t)c * F_DIM + k] = h;
    Wl[(size_t)c * F_DIM + k] = (_Float16)(v - (float)h);
}

// ---------------------------------------------------------------------------
// GEMM: out[M x 128] = A[M x 128] @ W[128 x 128]
// Block = 256 threads = 8 waves; one 16-row tile per block, wave w -> col tile w.
// A tile converted once to f16 hi/lo in LDS; B fragments are contiguous v16h
// loads from the pre-transposed/pre-converted weights. 3-product f16 split
// (Ah*Bh + Ah*Bl + Al*Bh) gives ~fp32 accuracy on V_WMMA_F32_16X16X32_F16.
// Fragment layouts: A and B both use contiguous k = 16*(lane/16) + s (wave32).
// ---------------------------------------------------------------------------
__global__ __launch_bounds__(256)
void gemm128_wmma(const float* __restrict__ A, const _Float16* __restrict__ Wh,
                  const _Float16* __restrict__ Wl, float* __restrict__ out, int M)
{
    __shared__ _Float16 sAh[16 * F_DIM];
    __shared__ _Float16 sAl[16 * F_DIM];

    const int tm = blockIdx.x;

    // ---- stage + convert A tile (16x128 fp32 -> f16 hi/lo), 8 elems/thread ----
    {
        int base = threadIdx.x * 8;            // [0, 2048)
        int r    = base >> 7;                  // tile row 0..15
        int k    = base & (F_DIM - 1);
        int row  = tm * 16 + r;
        int rowc = row < M ? row : (M - 1);    // clamp (stores are guarded)
        const float4* p = (const float4*)(A + (size_t)rowc * F_DIM + k);
        float4 v0 = p[0], v1 = p[1];
        float f[8] = {v0.x, v0.y, v0.z, v0.w, v1.x, v1.y, v1.z, v1.w};
        #pragma unroll
        for (int j = 0; j < 8; ++j) {
            _Float16 h = (_Float16)f[j];
            sAh[base + j] = h;
            sAl[base + j] = (_Float16)(f[j] - (float)h);
        }
    }
    __syncthreads();

    const int wave = threadIdx.x >> 5;         // col tile 0..7
    const int lane = threadIdx.x & 31;
    const bool hiH = lane >= 16;
    const int  l16 = lane & 15;
    const int  col = wave * 16 + l16;

    const _Float16* wch = Wh + (size_t)col * F_DIM;
    const _Float16* wcl = Wl + (size_t)col * F_DIM;
    const _Float16* arh = sAh + l16 * F_DIM;
    const _Float16* arl = sAl + l16 * F_DIM;

    v8f acc = {};

    #pragma unroll
    for (int kk = 0; kk < F_DIM; kk += 32) {
        // A fragment: lanes 0-15 hold K=kk..kk+15, lanes 16-31 K=kk+16..kk+31
        const int ka0 = kk + (hiH ? 16 : 0);
        const int ka1 = ka0 + 8;
        v8h a0h = *(const v8h*)(arh + ka0);
        v8h a1h = *(const v8h*)(arh + ka1);
        v8h a0l = *(const v8h*)(arl + ka0);
        v8h a1l = *(const v8h*)(arl + ka1);
        v16h ah, al;
        #pragma unroll
        for (int j = 0; j < 8; ++j) {
            ah[j] = a0h[j]; ah[8 + j] = a1h[j];
            al[j] = a0l[j]; al[8 + j] = a1l[j];
        }
        // B fragment: lanes 0-15 hold K=kk..kk+15, lanes 16-31 K=kk+16..kk+31
        const int kb = kk + (hiH ? 16 : 0);
        v16h bh = *(const v16h*)(wch + kb);
        v16h bl = *(const v16h*)(wcl + kb);

        acc = __builtin_amdgcn_wmma_f32_16x16x32_f16(false, ah, false, bh, (short)0, acc, false, false);
        acc = __builtin_amdgcn_wmma_f32_16x16x32_f16(false, ah, false, bl, (short)0, acc, false, false);
        acc = __builtin_amdgcn_wmma_f32_16x16x32_f16(false, al, false, bh, (short)0, acc, false, false);
    }

    // D store: VGPR j -> M = j (lanes 0-15) / j+8 (lanes 16-31), N = lane%16
    const int rowBase = tm * 16 + (hiH ? 8 : 0);
    #pragma unroll
    for (int j = 0; j < 8; ++j) {
        int r = rowBase + j;
        if (r < M) out[(size_t)r * F_DIM + col] = acc[j];
    }
}

// ---------------------------------------------------------------------------
// Edge scatter: one wave per edge; lane handles 4 consecutive features.
// agg[dst] += tmp[src] * normE[e]
// ---------------------------------------------------------------------------
__global__ __launch_bounds__(256)
void edge_scatter(const int* __restrict__ src, const int* __restrict__ dst,
                  const float* __restrict__ normE, const float* __restrict__ tmp,
                  float* __restrict__ agg, int E)
{
    int gid  = blockIdx.x * blockDim.x + threadIdx.x;
    int e    = gid >> 5;
    int lane = gid & 31;
    if (e >= E) return;
    int s = src[e], d = dst[e];
    float nm = normE[e];
    const float4 v = ((const float4*)(tmp + (size_t)s * F_DIM))[lane];
    float* ad = agg + (size_t)d * F_DIM + lane * 4;
    unsafeAtomicAdd(ad + 0, v.x * nm);
    unsafeAtomicAdd(ad + 1, v.y * nm);
    unsafeAtomicAdd(ad + 2, v.z * nm);
    unsafeAtomicAdd(ad + 3, v.w * nm);
}

// ---------------------------------------------------------------------------
// Fused self-loop message + bias + ReLU, in place on agg.
// ---------------------------------------------------------------------------
__global__ __launch_bounds__(256)
void fuse_bias_relu(const float* __restrict__ tmp, const float* __restrict__ dis,
                    const float* __restrict__ b, float* __restrict__ agg, int n)
{
    size_t i = (size_t)blockIdx.x * blockDim.x + threadIdx.x;
    if (i >= (size_t)n * F_DIM) return;
    int node = (int)(i >> 7);
    int f    = (int)(i & (F_DIM - 1));
    float di = dis[node];
    float v  = agg[i] + tmp[i] * di * di + b[f];
    agg[i]   = v > 0.0f ? v : 0.0f;
}

// ---------------------------------------------------------------------------
// global mean pool (sum + count via atomics)
// ---------------------------------------------------------------------------
__global__ __launch_bounds__(256)
void pool_sum(const float* __restrict__ h, const int* __restrict__ batch,
              float* __restrict__ sums, float* __restrict__ cnt, int n)
{
    int gid  = blockIdx.x * blockDim.x + threadIdx.x;
    int node = gid >> 5;
    int lane = gid & 31;
    if (node >= n) return;
    int g = batch[node];
    const float4 v = ((const float4*)(h + (size_t)node * F_DIM))[lane];
    float* p = sums + (size_t)g * F_DIM + lane * 4;
    unsafeAtomicAdd(p + 0, v.x);
    unsafeAtomicAdd(p + 1, v.y);
    unsafeAtomicAdd(p + 2, v.z);
    unsafeAtomicAdd(p + 3, v.w);
    if (lane == 0) unsafeAtomicAdd(&cnt[g], 1.0f);
}

// ---------------------------------------------------------------------------
// classifier: out[g][c] = (sums[g] / max(cnt,1)) . Wc[:,c] + bc[c]
// ---------------------------------------------------------------------------
__global__ __launch_bounds__(256)
void classify(const float* __restrict__ sums, const float* __restrict__ cnt,
              const float* __restrict__ Wc, const float* __restrict__ bc,
              float* __restrict__ out, int G, int C)
{
    int t = blockIdx.x * blockDim.x + threadIdx.x;
    if (t >= G * C) return;
    int g = t / C, c = t % C;
    float cn = cnt[g];
    cn = cn > 1.0f ? cn : 1.0f;
    float acc = 0.0f;
    #pragma unroll 4
    for (int f = 0; f < F_DIM; ++f)
        acc += sums[(size_t)g * F_DIM + f] * Wc[f * C + c];
    out[t] = acc / cn + bc[c];
}

// ---------------------------------------------------------------------------
// launch
// ---------------------------------------------------------------------------
extern "C" void kernel_launch(void* const* d_in, const int* in_sizes, int n_in,
                              void* d_out, int out_size, void* d_ws, size_t ws_size,
                              hipStream_t stream)
{
    const float* x     = (const float*)d_in[0];
    const int*   ei    = (const int*)d_in[1];
    const int*   batch = (const int*)d_in[2];
    const float* W1 = (const float*)d_in[3];
    const float* b1 = (const float*)d_in[4];
    const float* W2 = (const float*)d_in[5];
    const float* b2 = (const float*)d_in[6];
    const float* W3 = (const float*)d_in[7];
    const float* b3 = (const float*)d_in[8];
    const float* Wc = (const float*)d_in[9];
    const float* bc = (const float*)d_in[10];

    const int N = in_sizes[0] / F_DIM;
    const int E = in_sizes[1] / 2;
    const int C = in_sizes[9] / F_DIM;
    const int G = out_size / (C > 0 ? C : 1);

    const int* src = ei;
    const int* dst = ei + E;

    // workspace layout (float-sized slots, all 128-element aligned)
    float* ws   = (float*)d_ws;
    size_t NF   = (size_t)N * F_DIM;
    size_t npad = ((size_t)N + 255) & ~(size_t)255;
    size_t epad = ((size_t)E + 255) & ~(size_t)255;
    float* dis   = ws;                         // N   (deg first, then rsqrt in place)
    float* tmp   = dis + npad;                 // N*F (GEMM output)
    float* agg   = tmp + NF;                   // N*F (scatter target / activations)
    float* normE = agg + NF;                   // E   per-edge norm
    float* sums  = normE + epad;               // G*F
    float* cnt   = sums + (size_t)G * F_DIM;   // G
    _Float16* wth = (_Float16*)(cnt + ((size_t)G + 255 & ~(size_t)255)); // 128*128 f16
    _Float16* wtl = wth + (size_t)F_DIM * F_DIM;                          // 128*128 f16

    const int T = 256;
    // --- degree & normalization ---
    init_deg<<<(N + T - 1) / T, T, 0, stream>>>(dis, N);
    edge_deg<<<(E + T - 1) / T, T, 0, stream>>>(dst, dis, E);
    make_dis<<<(N + T - 1) / T, T, 0, stream>>>(dis, N);
    make_norm<<<(E + T - 1) / T, T, 0, stream>>>(src, dst, dis, normE, E);

    // --- 3 GCN layers ---
    const float* Ws[3] = {W1, W2, W3};
    const float* bs[3] = {b1, b2, b3};
    const float* in = x;
    for (int l = 0; l < 3; ++l) {
        conv_w<<<(F_DIM * F_DIM + T - 1) / T, T, 0, stream>>>(Ws[l], wth, wtl);
        gemm128_wmma<<<(N + 15) / 16, 256, 0, stream>>>(in, wth, wtl, tmp, N);
        hipMemsetAsync(agg, 0, NF * sizeof(float), stream);
        long long ethreads = (long long)E * 32;
        edge_scatter<<<(unsigned)((ethreads + T - 1) / T), T, 0, stream>>>(src, dst, normE, tmp, agg, E);
        fuse_bias_relu<<<(unsigned)((NF + T - 1) / T), T, 0, stream>>>(tmp, dis, bs[l], agg, N);
        in = agg;
    }

    // --- pool + classify ---
    hipMemsetAsync(sums, 0, ((size_t)G * F_DIM + G) * sizeof(float), stream);
    long long pthreads = (long long)N * 32;
    pool_sum<<<(unsigned)((pthreads + T - 1) / T), T, 0, stream>>>(agg, batch, sums, cnt, N);
    classify<<<1, 256, 0, stream>>>(sums, cnt, Wc, bc, (float*)d_out, G, C);
}
